// WaveLayer2D_11227044511879
// MI455X (gfx1250) — matrix operand order repaired and checked by reading, not compiled, not run on hardware
//
#include <hip/hip_runtime.h>
#include <hip/hip_bf16.h>
#include <stdint.h>

typedef float        v8f   __attribute__((ext_vector_type(8)));
typedef __bf16       v16bf __attribute__((ext_vector_type(16)));
typedef unsigned int u32x8 __attribute__((ext_vector_type(8)));

#define EPS_    1e-8f
#define LN_EPS_ 1e-5f

// ---------- helpers ----------
static __device__ __forceinline__ unsigned int f2bf(float f){
  unsigned int x = __float_as_uint(f);
  unsigned int r = x + 0x7fffu + ((x >> 16) & 1u);   // round-to-nearest-even
  return (r >> 16) & 0xffffu;
}

static __device__ __forceinline__ void polarf(float z, float g, float& cr, float& ci){
  float r = z / (g + EPS_);
  r = fminf(fmaxf(r, -1.0f + EPS_), 1.0f - EPS_);
  float s = sqrtf(fmaxf(1.0f - r * r, 0.0f));
  cr = g * r; ci = g * s;
}

static __device__ __forceinline__ float mod_absf(float a, float b, float c, float d){
  float pr = a * c - b * d;
  float pi = a * d + b * c;
  return sqrtf(pr * pr + pi * pi);
}

// ---------- pack x patches (32768 x 768) into bf16 WMMA A-fragment layout ----------
__global__ __launch_bounds__(256) void pack_A_kernel(const float* __restrict__ x,
                                                     unsigned int* __restrict__ Ap){
  int idx = blockIdx.x * 256 + threadIdx.x;          // 32768 * 384 threads
  int m = idx / 384;
  int j = idx - m * 384;                              // K pair index, d = 2j
  const float* src = x + (size_t)(m + (m >> 8) + 1) * 768 + (j << 1);
  unsigned int dw = f2bf(src[0]) | (f2bf(src[1]) << 16);
  int kt = j >> 4;
  int u  = j & 15;
  int v  = (u & 3) + ((u >> 3) << 2);
  int g  = (u >> 2) & 1;
  int lane = (m & 15) + (g << 4);
  int t = m >> 4;
  Ap[((size_t)(t * 24 + kt) * 32 + lane) * 8 + v] = dw;
}

// ---------- pack weight (768 x 768, row-major e,d) into bf16 B-fragment layout ----------
__global__ __launch_bounds__(256) void pack_W_kernel(const float* __restrict__ W,
                                                     unsigned int* __restrict__ Wp){
  int idx = blockIdx.x * 256 + threadIdx.x;           // 768 * 384 threads
  int n = idx / 384;
  int j = idx - n * 384;
  const float* src = W + (size_t)n * 768 + (j << 1);
  unsigned int dw = f2bf(src[0]) | (f2bf(src[1]) << 16);
  int kt = j >> 4;
  int u  = j & 15;
  int khalf = (u >> 2) & 1;
  int v  = (u & 3) + ((u >> 3) << 2);
  int lane = (n & 15) + (khalf << 4);
  int nt = n >> 4;
  Wp[((size_t)(nt * 24 + kt) * 32 + lane) * 8 + v] = dw;
}

// ---------- fused triple GEMM with async-LDS B staging ----------
// z_q = A @ W_q^T + b_q, q in {row,col,loc}
__global__ __launch_bounds__(256) void gemm3_kernel(
    const u32x8* __restrict__ Ap,
    const u32x8* __restrict__ Wp0, const u32x8* __restrict__ Wp1, const u32x8* __restrict__ Wp2,
    const float* __restrict__ b0, const float* __restrict__ b1, const float* __restrict__ b2,
    float* __restrict__ z0, float* __restrict__ z1, float* __restrict__ z2){
  __shared__ unsigned int sbuf[2][3072];              // double buffer: 2 x 12 KB of B frags
  const int tid  = threadIdx.x;
  const int lane = tid & 31;
  const int wave = tid >> 5;
  const int t  = blockIdx.x * 8 + wave;               // 16-row M tile [0,2048)
  const int nb = blockIdx.y * 4;                      // base 16-col N tile [0,48)

  const u32x8* wp[3] = {Wp0, Wp1, Wp2};

  // cooperative async copy of the 12 B fragments (12 KB) for k-tile kt -> sbuf[bsel]
  auto issueB = [&](int kt, int bsel){
    #pragma unroll
    for (int it = 0; it < 3; ++it){
      int unit = tid + (it << 8);                     // 0..767 units of 16B
      int f    = unit >> 6;                           // fragment 0..11 (q*4+n)
      int rem  = unit & 63;
      int l    = rem >> 1;                            // lane within fragment
      int hh   = rem & 1;                             // which 16B half
      int q    = f >> 2, n = f & 3;
      const char* src = (const char*)(wp[q] + ((size_t)((nb + n) * 24 + kt) * 32 + l))
                        + (hh << 4);
      unsigned lds = (unsigned)(uintptr_t)(&sbuf[bsel][0])
                     + (unsigned)((f << 10) + (l << 5) + (hh << 4));
      unsigned long long ga = (unsigned long long)(uintptr_t)src;
      asm volatile("global_load_async_to_lds_b128 %0, %1, off"
                   :: "v"(lds), "v"(ga) : "memory");
    }
  };

  v8f acc[3][4];
  #pragma unroll
  for (int q = 0; q < 3; ++q)
    #pragma unroll
    for (int n = 0; n < 4; ++n)
      #pragma unroll
      for (int i = 0; i < 8; ++i) acc[q][n][i] = 0.0f;

  issueB(0, 0);

  for (int kt = 0; kt < 24; ++kt){
    if (kt < 23){
      issueB(kt + 1, (kt + 1) & 1);
      asm volatile("s_wait_asynccnt 0x3" ::: "memory");   // kt's 3 async loads done
    } else {
      asm volatile("s_wait_asynccnt 0x0" ::: "memory");
    }
    __syncthreads();                                      // producers visible to all waves

    u32x8 av = Ap[(size_t)(t * 24 + kt) * 32 + lane];
    __builtin_prefetch(Ap + (size_t)(t * 24 + kt + 1) * 32 + lane, 0, 3);
    v16bf a = __builtin_bit_cast(v16bf, av);

    const int bsel = kt & 1;
    #pragma unroll
    for (int q = 0; q < 3; ++q){
      #pragma unroll
      for (int n = 0; n < 4; ++n){
        u32x8 bv = *(const u32x8*)&sbuf[bsel][((q * 4 + n) << 8) + (lane << 3)];
        v16bf b = __builtin_bit_cast(v16bf, bv);
        acc[q][n] = __builtin_amdgcn_wmma_f32_16x16x32_bf16(
            false, a, false, b, (short)0, acc[q][n], false, false);
      }
    }
    __syncthreads();                                      // safe to overwrite this buffer
  }

  const float* bias[3] = {b0, b1, b2};
  float* zout[3] = {z0, z1, z2};
  const int row0 = t * 16 + ((lane >> 4) << 3);
  const int col0 = (nb << 4) + (lane & 15);
  #pragma unroll
  for (int q = 0; q < 3; ++q){
    #pragma unroll
    for (int n = 0; n < 4; ++n){
      const int col = col0 + (n << 4);
      const float bb = bias[q][col];
      #pragma unroll
      for (int r = 0; r < 8; ++r)
        zout[q][(size_t)(row0 + r) * 768 + col] = acc[q][n][r] + bb;
    }
  }
}

// ---------- S_row[b,h,e] = sum_w z_row^2 ----------
__global__ __launch_bounds__(256) void rownorm_kernel(const float* __restrict__ z,
                                                      float* __restrict__ S){
  int bh = blockIdx.x;
  size_t base = (size_t)bh * 16 * 768;
  for (int e = threadIdx.x; e < 768; e += 256){
    float acc = 0.f;
    #pragma unroll
    for (int w = 0; w < 16; ++w){
      float v = z[base + (size_t)w * 768 + e];
      acc += v * v;
    }
    S[(size_t)bh * 768 + e] = acc;
  }
}

// ---------- S_col[b,w,e] = sum_h z_col^2 ----------
__global__ __launch_bounds__(256) void colnorm_kernel(const float* __restrict__ z,
                                                      float* __restrict__ S){
  int bw = blockIdx.x;
  int b = bw >> 4, w = bw & 15;
  size_t base = (size_t)b * 256 * 768 + (size_t)w * 768;
  for (int e = threadIdx.x; e < 768; e += 256){
    float acc = 0.f;
    #pragma unroll
    for (int h = 0; h < 16; ++h){
      float v = z[base + (size_t)h * 16 * 768 + e];
      acc += v * v;
    }
    S[(size_t)bw * 768 + e] = acc;
  }
}

// ---------- per-token: wave ops + fusion + residual + LayerNorm ----------
__global__ __launch_bounds__(256) void fusion_kernel(
    const float* __restrict__ x,
    const float* __restrict__ zr, const float* __restrict__ zc, const float* __restrict__ zl,
    const float* __restrict__ Sr, const float* __restrict__ Sc,
    const float* __restrict__ fw, const float* __restrict__ gamma, const float* __restrict__ beta,
    float* __restrict__ out){
  __shared__ float2 red[256];
  const int tid = threadIdx.x;
  const int tok = blockIdx.x;
  const int b = tok >> 8, p = tok & 255;
  const int h = p >> 4,  w = p & 15;
  const int hm = (h + 15) & 15, wm = (w + 15) & 15;
  const size_t img    = (size_t)b * 256 * 768;
  const size_t base   = img + (size_t)p * 768;
  const size_t baseWm = img + (size_t)(h  * 16 + wm) * 768;
  const size_t baseHm = img + (size_t)(hm * 16 + w ) * 768;
  const size_t baseSh = img + (size_t)(hm * 16 + wm) * 768;
  const size_t srB = (size_t)(b * 16 + h) * 768;
  const size_t scB = (size_t)(b * 16 + w) * 768;
  const size_t obase = (size_t)(b * 257 + 1 + p) * 768;

  float f0 = fw[0], f1 = fw[1], f2 = fw[2];
  float mx = fmaxf(f0, fmaxf(f1, f2));
  float e0 = __expf(f0 - mx), e1 = __expf(f1 - mx), e2 = __expf(f2 - mx);
  float inv_s = 1.0f / (e0 + e1 + e2);
  float w0 = e0 * inv_s, w1 = e1 * inv_s, w2 = e2 * inv_s;

  float lv[3];
  float s = 0.f, s2 = 0.f;
  #pragma unroll
  for (int i = 0; i < 3; ++i){
    int e = tid + i * 256;
    float cr, ci, nr, ni;
    float g = sqrtf(Sr[srB + e] + EPS_);
    polarf(zr[base + e],   g, cr, ci);
    polarf(zr[baseWm + e], g, nr, ni);
    float rowv = mod_absf(cr, ci, nr, ni);
    g = sqrtf(Sc[scB + e] + EPS_);
    polarf(zc[base + e],   g, cr, ci);
    polarf(zc[baseHm + e], g, nr, ni);
    float colv = mod_absf(cr, ci, nr, ni);
    float win = 0.f;
    #pragma unroll
    for (int dh = -1; dh <= 1; ++dh){
      int hh = min(max(h + dh, 0), 15);
      #pragma unroll
      for (int dw = -1; dw <= 1; ++dw){
        int ww = min(max(w + dw, 0), 15);
        float v = zl[img + (size_t)(hh * 16 + ww) * 768 + e];
        win += v * v;
      }
    }
    g = sqrtf(win + EPS_);
    polarf(zl[base + e],   g, cr, ci);
    polarf(zl[baseSh + e], g, nr, ni);
    float locv = mod_absf(cr, ci, nr, ni);

    float val = w0 * rowv + w1 * colv + w2 * locv + x[obase + e];
    lv[i] = val;
    s += val; s2 += val * val;
  }
  red[tid] = make_float2(s, s2);
  __syncthreads();
  for (int o = 128; o > 0; o >>= 1){
    if (tid < o){ red[tid].x += red[tid + o].x; red[tid].y += red[tid + o].y; }
    __syncthreads();
  }
  float mu   = red[0].x * (1.0f / 768.0f);
  float var  = red[0].y * (1.0f / 768.0f) - mu * mu;
  float rinv = rsqrtf(var + LN_EPS_);
  #pragma unroll
  for (int i = 0; i < 3; ++i){
    int e = tid + i * 256;
    out[obase + e] = (lv[i] - mu) * rinv * gamma[e] + beta[e];
  }
}

// ---------- CLS token: mean over patch_out + residual + LayerNorm ----------
__global__ __launch_bounds__(256) void cls_kernel(const float* __restrict__ x,
                                                  const float* __restrict__ gamma,
                                                  const float* __restrict__ beta,
                                                  float* __restrict__ out){
  __shared__ float2 red[256];
  const int tid = threadIdx.x;
  const int b = blockIdx.x;
  const size_t rowbase = (size_t)(b * 257) * 768;
  float lv[3];
  float s = 0.f, s2 = 0.f;
  #pragma unroll
  for (int i = 0; i < 3; ++i){
    int e = tid + i * 256;
    float acc = 0.f;
    for (int p = 0; p < 256; ++p)
      acc += out[rowbase + (size_t)(1 + p) * 768 + e];
    float val = x[rowbase + e] + acc * (1.0f / 256.0f);
    lv[i] = val;
    s += val; s2 += val * val;
  }
  red[tid] = make_float2(s, s2);
  __syncthreads();
  for (int o = 128; o > 0; o >>= 1){
    if (tid < o){ red[tid].x += red[tid + o].x; red[tid].y += red[tid + o].y; }
    __syncthreads();
  }
  float mu   = red[0].x * (1.0f / 768.0f);
  float var  = red[0].y * (1.0f / 768.0f) - mu * mu;
  float rinv = rsqrtf(var + LN_EPS_);
  #pragma unroll
  for (int i = 0; i < 3; ++i){
    int e = tid + i * 256;
    out[rowbase + e] = (lv[i] - mu) * rinv * gamma[e] + beta[e];
  }
}

// ---------- launch ----------
extern "C" void kernel_launch(void* const* d_in, const int* in_sizes, int n_in,
                              void* d_out, int out_size, void* d_ws, size_t ws_size,
                              hipStream_t stream){
  (void)in_sizes; (void)n_in; (void)out_size; (void)ws_size;
  const float* x     = (const float*)d_in[0];
  const float* W_row = (const float*)d_in[1];
  const float* b_row = (const float*)d_in[2];
  const float* W_col = (const float*)d_in[3];
  const float* b_col = (const float*)d_in[4];
  const float* W_loc = (const float*)d_in[5];
  const float* b_loc = (const float*)d_in[6];
  const float* fw    = (const float*)d_in[7];
  const float* gamma = (const float*)d_in[8];
  const float* beta  = (const float*)d_in[9];
  float* out = (float*)d_out;

  char* ws = (char*)d_ws;
  size_t off = 0;
  auto take = [&](size_t bytes)->char*{
    char* p = ws + off;
    off += (bytes + 255) & ~(size_t)255;
    return p;
  };
  unsigned int* Ap  = (unsigned int*)take((size_t)32768 * 768 * 2);
  unsigned int* Wp0 = (unsigned int*)take((size_t)768 * 768 * 2);
  unsigned int* Wp1 = (unsigned int*)take((size_t)768 * 768 * 2);
  unsigned int* Wp2 = (unsigned int*)take((size_t)768 * 768 * 2);
  float* z0 = (float*)take((size_t)32768 * 768 * 4);
  float* z1 = (float*)take((size_t)32768 * 768 * 4);
  float* z2 = (float*)take((size_t)32768 * 768 * 4);
  float* Sr = (float*)take((size_t)2048 * 768 * 4);
  float* Sc = (float*)take((size_t)2048 * 768 * 4);

  pack_A_kernel<<<49152, 256, 0, stream>>>(x, Ap);
  pack_W_kernel<<<1152, 256, 0, stream>>>(W_row, Wp0);
  pack_W_kernel<<<1152, 256, 0, stream>>>(W_col, Wp1);
  pack_W_kernel<<<1152, 256, 0, stream>>>(W_loc, Wp2);

  gemm3_kernel<<<dim3(256, 12), 256, 0, stream>>>(
      (const u32x8*)Ap, (const u32x8*)Wp0, (const u32x8*)Wp1, (const u32x8*)Wp2,
      b_row, b_col, b_loc, z0, z1, z2);

  rownorm_kernel<<<2048, 256, 0, stream>>>(z0, Sr);
  colnorm_kernel<<<2048, 256, 0, stream>>>(z1, Sc);

  fusion_kernel<<<32768, 256, 0, stream>>>(x, z0, z1, z2, Sr, Sc, fw, gamma, beta, out);
  cls_kernel<<<128, 256, 0, stream>>>(x, gamma, beta, out);
}
